// VectorQuantizer_56813827391976
// MI455X (gfx1250) — compile-verified, matrix operand order checked by reading, not run on hardware
//
#include <hip/hip_runtime.h>
#include <hip/hip_bf16.h>
#include <stdint.h>

// ---------------------------------------------------------------------------
// VQ-VAE vector quantizer for MI455X (gfx1250, wave32, WMMA bf16)
// ---------------------------------------------------------------------------

typedef __bf16 bf16_t;
typedef bf16_t bf16x8  __attribute__((ext_vector_type(8)));
typedef bf16_t bf16x16 __attribute__((ext_vector_type(16)));
typedef float  f32x8   __attribute__((ext_vector_type(8)));
typedef int    v4i     __attribute__((ext_vector_type(4)));

#define VQ_D       256   // latent / codebook dim (reference fixed)
#define VQ_MTILE   64    // rows per block (4 waves x 16 rows)
#define VQ_CTILE   16    // codes per LDS tile
#define LDS_STRIDE 264   // 256 halves + 8 pad halves (16B) to spread banks

#if defined(__HIP_DEVICE_COMPILE__) && \
    __has_builtin(__builtin_amdgcn_global_load_async_to_lds_b128) && \
    __has_builtin(__builtin_amdgcn_s_wait_asynccnt)
#define VQ_ASYNC 1
#else
#define VQ_ASYNC 0
#endif

#if VQ_ASYNC
typedef __attribute__((address_space(1))) v4i as1_v4i;  // global 128-bit chunk
typedef __attribute__((address_space(3))) v4i as3_v4i;  // LDS 128-bit chunk
#endif

__device__ __forceinline__ bf16_t f2bf(float f) {
  // round-to-nearest-even f32 -> bf16
  unsigned u = __builtin_bit_cast(unsigned, f);
  unsigned r = u + 0x7FFFu + ((u >> 16) & 1u);
  unsigned short hs = (unsigned short)(r >> 16);
  return __builtin_bit_cast(bf16_t, hs);
}

// ---------------------------------------------------------------------------
// Kernel 1: codebook -> bf16 copy + per-row squared norms
// ---------------------------------------------------------------------------
__global__ __launch_bounds__(256) void vq_prep(const float* __restrict__ embed,
                                               bf16_t* __restrict__ cb,
                                               float* __restrict__ cnorm) {
  const int c = blockIdx.x;
  const int t = threadIdx.x;
  const float w = embed[(size_t)c * VQ_D + t];
  cb[(size_t)c * VQ_D + t] = f2bf(w);
  __shared__ float red[256];
  red[t] = w * w;
  __syncthreads();
  #pragma unroll
  for (int s = 128; s > 0; s >>= 1) {
    if (t < s) red[t] += red[t + s];
    __syncthreads();
  }
  if (t == 0) cnorm[c] = red[0];
}

// ---------------------------------------------------------------------------
// LDS tile stage: 16 codes x 256 bf16 (8KB) as 512 x 16B chunks, 128 threads
// ---------------------------------------------------------------------------
__device__ __forceinline__ void issue_tile(const bf16_t* __restrict__ cb,
                                           int codebase, bf16_t* dst, int tid) {
  #pragma unroll
  for (int i = 0; i < 4; ++i) {
    const int chunk = tid + i * 128;
    const int row   = chunk >> 5;         // 0..15
    const int off   = (chunk & 31) * 8;   // halves, 16B granules
    const bf16_t* g = cb + (size_t)(codebase + row) * VQ_D + off;
    bf16_t*       l = dst + row * LDS_STRIDE + off;
#if VQ_ASYNC
    __builtin_amdgcn_global_load_async_to_lds_b128(
        (as1_v4i*)(bf16_t*)g, (as3_v4i*)l, 0, 0);
#else
    *(uint4*)l = *(const uint4*)g;
#endif
  }
}

// ---------------------------------------------------------------------------
// Kernel 2: distance GEMM via WMMA bf16 + running argmin + gather/output
// ---------------------------------------------------------------------------
__global__ __launch_bounds__(128) void vq_main(const float* __restrict__ lat,
                                               const float* __restrict__ embed,
                                               const bf16_t* __restrict__ cb,
                                               const float* __restrict__ cnorm,
                                               float* __restrict__ qout,
                                               float* __restrict__ inds,
                                               float* __restrict__ partials,
                                               int K) {
  __shared__ __align__(16) bf16_t tile[2][VQ_CTILE][LDS_STRIDE];

  const int tid  = threadIdx.x;
  const int wave = tid >> 5;
  const int lane = tid & 31;
  const int m    = lane & 15;   // row within A-frag / code column within tile
  const int h    = lane >> 4;   // half selector for 16-bit fragment layout
  const int rowbase = blockIdx.x * VQ_MTILE + wave * 16;

  // --- Preload A fragments: 16 rows x 256 d in CDNA5 16-bit A layout.
  // Lane (m,h) holds, per 32-wide k-chunk: elems 0..7 = d[kk+8h .. +7],
  // elems 8..15 = d[kk+16+8h .. +7].
  const float* xrow = lat + (size_t)(rowbase + m) * VQ_D + 8 * h;
  bf16x16 A[8];
  #pragma unroll
  for (int c = 0; c < 8; ++c) {
    const float4 f0 = *(const float4*)(xrow + c * 32);
    const float4 f1 = *(const float4*)(xrow + c * 32 + 4);
    const float4 f2 = *(const float4*)(xrow + c * 32 + 16);
    const float4 f3 = *(const float4*)(xrow + c * 32 + 20);
    bf16x16 a;
    a[0]  = f2bf(f0.x); a[1]  = f2bf(f0.y); a[2]  = f2bf(f0.z); a[3]  = f2bf(f0.w);
    a[4]  = f2bf(f1.x); a[5]  = f2bf(f1.y); a[6]  = f2bf(f1.z); a[7]  = f2bf(f1.w);
    a[8]  = f2bf(f2.x); a[9]  = f2bf(f2.y); a[10] = f2bf(f2.z); a[11] = f2bf(f2.w);
    a[12] = f2bf(f3.x); a[13] = f2bf(f3.y); a[14] = f2bf(f3.z); a[15] = f2bf(f3.w);
    A[c] = a;
  }

  float best[8];
  int   bidx[8];
  #pragma unroll
  for (int j = 0; j < 8; ++j) { best[j] = 3.4e38f; bidx[j] = 0x7fffffff; }

  const int T = K / VQ_CTILE;
  issue_tile(cb, 0, &tile[0][0][0], tid);

  for (int t = 0; t < T; ++t) {
    const int buf = t & 1;
#if VQ_ASYNC
    __builtin_amdgcn_s_wait_asynccnt(0);
#endif
    __syncthreads();  // tile t resident; all waves done with buf^1
    if (t + 1 < T) issue_tile(cb, (t + 1) * VQ_CTILE, &tile[buf ^ 1][0][0], tid);

    const int   codebase = t * VQ_CTILE;
    const float cn       = cnorm[codebase + m];
    f32x8 acc = {0.f, 0.f, 0.f, 0.f, 0.f, 0.f, 0.f, 0.f};
    const bf16_t* trow = &tile[buf][m][0];

    // Software-pipelined B fragments: issue ds_loads for chunk c+1 before
    // the WMMA consuming chunk c, so DS runs ahead of the matrix pipe and
    // the per-WMMA wait drops from dscnt==0 to dscnt<=2.
    bf16x8 pb0 = *(const bf16x8*)(trow + 8 * h);         // ds_read_b128
    bf16x8 pb1 = *(const bf16x8*)(trow + 8 * h + 16);    // ds_read_b128
    #pragma unroll
    for (int c = 0; c < 8; ++c) {
      const bf16x8 cb0 = pb0;
      const bf16x8 cb1 = pb1;
      if (c < 7) {
        pb0 = *(const bf16x8*)(trow + (c + 1) * 32 + 8 * h);
        pb1 = *(const bf16x8*)(trow + (c + 1) * 32 + 8 * h + 16);
      }
      bf16x16 bb = __builtin_shufflevector(cb0, cb1,
          0, 1, 2, 3, 4, 5, 6, 7, 8, 9, 10, 11, 12, 13, 14, 15);
      acc = __builtin_amdgcn_wmma_f32_16x16x32_bf16(
          false, A[c], false, bb, (short)0, acc, false, false);
    }

    const int code = codebase + m;
    #pragma unroll
    for (int j = 0; j < 8; ++j) {
      const float s = cn - 2.0f * acc[j];   // ||e||^2 - 2<x,e>
      if (s < best[j]) { best[j] = s; bidx[j] = code; }
    }
  }

  // --- Cross-lane argmin: reduce over the 16 code columns per half-group.
  #pragma unroll
  for (int j = 0; j < 8; ++j) {
    #pragma unroll
    for (int d = 1; d < 16; d <<= 1) {
      const float ob = __shfl_xor(best[j], d, 16);
      const int   oi = __shfl_xor(bidx[j], d, 16);
      if (ob < best[j] || (ob == best[j] && oi < bidx[j])) {
        best[j] = ob; bidx[j] = oi;
      }
    }
  }
  // Now every lane of each 16-lane group holds the winners:
  // group h, slot j -> row (j + 8h), best code index in bidx[j].

  // Write indices (as float, matching concatenated f32 output layout).
  if (m == 0) {
    #pragma unroll
    for (int j = 0; j < 8; ++j)
      inds[rowbase + 8 * h + j] = (float)bidx[j];
  }

  // Broadcast the 16 winning codes to the whole wave.
  int codes[16];
  #pragma unroll
  for (int r = 0; r < 16; ++r)
    codes[r] = __shfl(bidx[r & 7], (r < 8) ? 0 : 16, 32);

  // --- Gather f32 codebook rows, emit quantized (== STE output), loss partial.
  float lsum = 0.f;
  #pragma unroll
  for (int r = 0; r < 16; ++r) {
    const int row = rowbase + r;
    const float* e = embed + (size_t)codes[r] * VQ_D + lane * 8;
    const float* x = lat   + (size_t)row      * VQ_D + lane * 8;
    const float4 e0 = *(const float4*)(e);
    const float4 e1 = *(const float4*)(e + 4);
    const float4 x0 = *(const float4*)(x);
    const float4 x1 = *(const float4*)(x + 4);
    *(float4*)(qout + (size_t)row * VQ_D + lane * 8)     = e0;
    *(float4*)(qout + (size_t)row * VQ_D + lane * 8 + 4) = e1;
    float d0 = e0.x - x0.x, d1 = e0.y - x0.y, d2 = e0.z - x0.z, d3 = e0.w - x0.w;
    float d4 = e1.x - x1.x, d5 = e1.y - x1.y, d6 = e1.z - x1.z, d7 = e1.w - x1.w;
    lsum += d0 * d0 + d1 * d1 + d2 * d2 + d3 * d3 +
            d4 * d4 + d5 * d5 + d6 * d6 + d7 * d7;
  }
  #pragma unroll
  for (int d = 16; d > 0; d >>= 1) lsum += __shfl_xor(lsum, d, 32);
  if (lane == 0) partials[blockIdx.x * 4 + wave] = lsum;
}

// ---------------------------------------------------------------------------
// Kernel 3: deterministic fixed-order loss reduction
// ---------------------------------------------------------------------------
__global__ __launch_bounds__(256) void vq_finalize(const float* __restrict__ partials,
                                                   int n, float* __restrict__ loss,
                                                   float scale) {
  __shared__ float red[256];
  const int t = threadIdx.x;
  float v = 0.f;
  for (int i = t; i < n; i += 256) v += partials[i];
  red[t] = v;
  __syncthreads();
  #pragma unroll
  for (int s = 128; s > 0; s >>= 1) {
    if (t < s) red[t] += red[t + s];
    __syncthreads();
  }
  if (t == 0) loss[0] = red[0] * scale;
}

// ---------------------------------------------------------------------------
extern "C" void kernel_launch(void* const* d_in, const int* in_sizes, int n_in,
                              void* d_out, int out_size, void* d_ws, size_t ws_size,
                              hipStream_t stream) {
  const float* lat   = (const float*)d_in[0];   // [N, 256] flattened latents
  const float* embed = (const float*)d_in[1];   // [K, 256] codebook
  const int N = in_sizes[0] / VQ_D;             // 32768
  const int K = in_sizes[1] / VQ_D;             // 4096

  char* ws = (char*)d_ws;
  bf16_t* cb      = (bf16_t*)ws;                                   // K*256 bf16
  float*  cnorm   = (float*)(ws + (size_t)K * VQ_D * sizeof(bf16_t));
  float*  parts   = cnorm + K;

  float* qout = (float*)d_out;                  // [N,256] quantized (== STE out)
  float* loss = qout + (size_t)N * VQ_D;        // scalar vq_loss
  float* inds = loss + 1;                       // [N] indices (as f32)

  const int nblocks = N / VQ_MTILE;             // 512
  const int nparts  = nblocks * 4;

  vq_prep<<<K, 256, 0, stream>>>(embed, cb, cnorm);
  vq_main<<<nblocks, 128, 0, stream>>>(lat, embed, cb, cnorm, qout, inds, parts, K);
  // vq_loss = 1.25 * mean((q - x)^2): commitment==embedding numerically; BETA=0.25
  vq_finalize<<<1, 256, 0, stream>>>(parts, nparts, loss, 1.25f / ((float)N * (float)VQ_D));
}
